// GAT_65970697666939
// MI455X (gfx1250) — matrix operand order repaired and checked by reading, not compiled, not run on hardware
//
#include <hip/hip_runtime.h>
#include <hip/hip_bf16.h>
#include <math.h>

// ---------------------------------------------------------------------------
// Types for WMMA (gfx1250, wave32)
// ---------------------------------------------------------------------------
typedef __attribute__((ext_vector_type(16))) _Float16 v16h;
typedef __attribute__((ext_vector_type(8)))  _Float16 v8h;
typedef __attribute__((ext_vector_type(8)))  float    v8f;

// Monotonic order-preserving float<->uint encoding for atomicMax on floats.
__device__ __forceinline__ unsigned f2o(float f) {
  unsigned u = __float_as_uint(f);
  return (u & 0x80000000u) ? ~u : (u | 0x80000000u);
}
__device__ __forceinline__ float o2f(unsigned u) {
  return (u & 0x80000000u) ? __uint_as_float(u & 0x7fffffffu)
                           : __uint_as_float(~u);
}

// ---------------------------------------------------------------------------
// fp32 -> fp16 conversion of node features
// ---------------------------------------------------------------------------
__global__ void k_cvt_x(const float* __restrict__ x, _Float16* __restrict__ xh, int total) {
  int tid = blockIdx.x * blockDim.x + threadIdx.x;
  if (tid < total) xh[tid] = (_Float16)x[tid];
}

// Convert + transpose weights: W0 (128x256) -> w0t (256x128), W1 (256x32) -> w1t (32x256)
__global__ void k_cvt_w(const float* __restrict__ W0, const float* __restrict__ W1,
                        _Float16* __restrict__ w0t, _Float16* __restrict__ w1t) {
  int tid = blockIdx.x * blockDim.x + threadIdx.x;
  const int n0 = 128 * 256;
  const int n1 = 256 * 32;
  if (tid < n0) {
    int k = tid / 256, n = tid % 256;
    w0t[n * 128 + k] = (_Float16)W0[tid];
  } else if (tid < n0 + n1) {
    int t = tid - n0;
    int k = t / 32, n = t % 32;
    w1t[n * 256 + k] = (_Float16)W1[t];
  }
}

// ---------------------------------------------------------------------------
// WMMA GEMM:  C[M x Nout] = A[M x K] (f16, row-major) * Bt[Nout x K] (f16, B^T)
// Each wave32 computes a 16x32 strip: one A fragment, two 16x16 WMMA tiles.
// A rows are clamped (not masked) so EXEC stays all-ones through the loop and
// no zero-fill/exec-save churn appears in the hot path; clamped rows produce
// garbage accumulators that are simply never stored.
// VGPR layouts per CDNA5 ISA 7.12.2 (16-bit A 16x32, 16-bit B 32x16).
// ---------------------------------------------------------------------------
__global__ void k_gemm_wmma(const _Float16* __restrict__ A,
                            const _Float16* __restrict__ Bt,
                            float* __restrict__ C,
                            int M, int K, int Nout) {
  int lane = threadIdx.x & 31;
  int wave = threadIdx.x >> 5;
  int tilesN2 = Nout >> 5;                    // 32-wide N strips
  int tilesM  = (M + 15) >> 4;
  int tile = blockIdx.x * (blockDim.x >> 5) + wave;
  if (tile >= tilesM * tilesN2) return;       // uniform per wave
  int tm  = tile / tilesN2;
  int tn2 = tile - tm * tilesN2;

  int l15  = lane & 15;
  int hgrp = lane >> 4;                       // half-wave select
  int row  = tm * 16 + l15;
  if (row >= M) row = M - 1;                  // clamp: keep EXEC full
  int col0 = tn2 * 32 + l15;
  int col1 = col0 + 16;
  int akg  = hgrp * 8;                        // A: k-subgroup (0 or 8) within 32
  int bkg  = hgrp * 8;                        // B: k-subgroup (0 or 8) within 32

  const _Float16* ap  = A  + (size_t)row  * K + akg;
  const _Float16* bp0 = Bt + (size_t)col0 * K + bkg;
  const _Float16* bp1 = Bt + (size_t)col1 * K + bkg;

  v8f acc0 = {}, acc1 = {};
  for (int k0 = 0; k0 < K; k0 += 32) {
    v8h alo = *(const v8h*)(ap + k0);         // K = k0+akg .. +7
    v8h ahi = *(const v8h*)(ap + k0 + 16);    // K = k0+16+akg .. +7
    v16h a = __builtin_shufflevector(alo, ahi, 0,1,2,3,4,5,6,7,8,9,10,11,12,13,14,15);

    v8h b0lo = *(const v8h*)(bp0 + k0);
    v8h b0hi = *(const v8h*)(bp0 + k0 + 16);
    v16h b0 = __builtin_shufflevector(b0lo, b0hi, 0,1,2,3,4,5,6,7,8,9,10,11,12,13,14,15);
    acc0 = __builtin_amdgcn_wmma_f32_16x16x32_f16(false, a, false, b0, (short)0, acc0, false, false);

    v8h b1lo = *(const v8h*)(bp1 + k0);
    v8h b1hi = *(const v8h*)(bp1 + k0 + 16);
    v16h b1 = __builtin_shufflevector(b1lo, b1hi, 0,1,2,3,4,5,6,7,8,9,10,11,12,13,14,15);
    acc1 = __builtin_amdgcn_wmma_f32_16x16x32_f16(false, a, false, b1, (short)0, acc1, false, false);
  }

  int mbase = tm * 16 + hgrp * 8;
#pragma unroll
  for (int v = 0; v < 8; ++v) {
    int m = mbase + v;
    if (m < M) {
      C[(size_t)m * Nout + col0] = acc0[v];   // D layout: VGPR v -> row mbase+v
      C[(size_t)m * Nout + col1] = acc1[v];
    }
  }
}

// ---------------------------------------------------------------------------
// Per-(node, head) attention logits:  el = sum_d f*al,  er = sum_d f*ar
// ---------------------------------------------------------------------------
__global__ void k_elsum(const float* __restrict__ f, const float* __restrict__ al,
                        const float* __restrict__ ar, float* __restrict__ el,
                        float* __restrict__ er, int Nn, int H) {
  int tid = blockIdx.x * blockDim.x + threadIdx.x;
  if (tid >= Nn * H) return;
  int n = tid / H, h = tid - n * H;
  int C = H * 32;
  const float* fr = f + (size_t)n * C + h * 32;
  float sl = 0.f, sr = 0.f;
#pragma unroll
  for (int d = 0; d < 32; ++d) {
    float v = fr[d];
    sl += v * al[h * 32 + d];
    sr += v * ar[h * 32 + d];
  }
  el[tid] = sl;
  er[tid] = sr;
}

// Pass A: e = leaky_relu(el[src]+er[dst]); segment-max via encoded atomicMax
__global__ void k_edge_a(const float* __restrict__ el, const float* __restrict__ er,
                         const int* __restrict__ src, const int* __restrict__ dst,
                         float* __restrict__ e, unsigned* __restrict__ m, int E, int H) {
  int tid = blockIdx.x * blockDim.x + threadIdx.x;
  if (tid >= E * H) return;
  int edge = tid / H, h = tid - edge * H;
  int s = src[edge], t = dst[edge];
  float v = el[s * H + h] + er[t * H + h];
  v = v > 0.f ? v : 0.2f * v;
  e[tid] = v;
  atomicMax(&m[t * H + h], f2o(v));
}

// Pass B: a = exp(e - max[dst]); denom[dst] += a
__global__ void k_edge_b(float* __restrict__ e, const unsigned* __restrict__ m,
                         float* __restrict__ den, const int* __restrict__ dst, int E, int H) {
  int tid = blockIdx.x * blockDim.x + threadIdx.x;
  if (tid >= E * H) return;
  int edge = tid / H, h = tid - edge * H;
  int t = dst[edge];
  float a = expf(e[tid] - o2f(m[t * H + h]));
  e[tid] = a;
  atomicAdd(&den[t * H + h], a);
}

// Pass C: out[dst] += (a/denom[dst]) * f[src]   (one wave32 per edge, lane = dim)
__global__ void k_edge_c(const float* __restrict__ e, const float* __restrict__ den,
                         const float* __restrict__ f, const int* __restrict__ src,
                         const int* __restrict__ dst, float* __restrict__ out, int E, int H) {
  int tid = blockIdx.x * blockDim.x + threadIdx.x;
  int edge = tid >> 5;
  int lane = tid & 31;
  if (edge >= E) return;
  int s = src[edge], t = dst[edge];
  int C = H * 32;
  const float* fr = f + (size_t)s * C;
  float* orow = out + (size_t)t * C;
  __builtin_prefetch(fr + lane, 0, 1);          // global_prefetch on the gathered row
  for (int h = 0; h < H; ++h) {
    float w = e[edge * H + h] / den[t * H + h];
    atomicAdd(&orow[h * 32 + lane], w * fr[h * 32 + lane]);
  }
}

// h = elu(out0 + b0), stored directly as f16 for the next GEMM
__global__ void k_elu(const float* __restrict__ out0, const float* __restrict__ b0,
                      _Float16* __restrict__ hb, int total) {
  int tid = blockIdx.x * blockDim.x + threadIdx.x;
  if (tid >= total) return;
  float v = out0[tid] + b0[tid & 255];
  v = v > 0.f ? v : (expf(v) - 1.f);
  hb[tid] = (_Float16)v;
}

// ---------------------------------------------------------------------------
// Link predictor: out[i] = P3( relu(P2( relu(P1( za*zb )))) ), b1 folded in.
// One thread per output row; fully-unrolled 32-wide register math.
// ---------------------------------------------------------------------------
__global__ void k_pred(const float* __restrict__ h, const float* __restrict__ b1,
                       const float* __restrict__ P1, const float* __restrict__ pb1,
                       const float* __restrict__ P2, const float* __restrict__ pb2,
                       const float* __restrict__ P3, const float* __restrict__ pb3,
                       const int* __restrict__ negp, int Nn,
                       float* __restrict__ out, int rows) {
  int i = blockIdx.x * blockDim.x + threadIdx.x;
  if (i >= rows) return;
  int neg = negp[0];
  int ne = Nn / (neg + 2);
  int ra, rb;
  if (i < ne) { ra = i; rb = ne + i; }
  else        { int j = i - ne; ra = j % ne; rb = 2 * ne + j; }

  float z[32];
#pragma unroll
  for (int d = 0; d < 32; ++d)
    z[d] = (h[(size_t)ra * 32 + d] + b1[d]) * (h[(size_t)rb * 32 + d] + b1[d]);

  float z1[32];
#pragma unroll
  for (int o = 0; o < 32; ++o) {
    float s = pb1[o];
#pragma unroll
    for (int d = 0; d < 32; ++d) s += z[d] * P1[d * 32 + o];
    z1[o] = fmaxf(s, 0.f);
  }
  float z2[32];
#pragma unroll
  for (int o = 0; o < 32; ++o) {
    float s = pb2[o];
#pragma unroll
    for (int d = 0; d < 32; ++d) s += z1[d] * P2[d * 32 + o];
    z2[o] = fmaxf(s, 0.f);
  }
  float s = pb3[0];
#pragma unroll
  for (int d = 0; d < 32; ++d) s += z2[d] * P3[d];
  out[i] = s;
}

// ---------------------------------------------------------------------------
static inline int cdiv(long long a, long long b) { return (int)((a + b - 1) / b); }
static inline size_t aln(size_t x) { return (x + 255) & ~(size_t)255; }

extern "C" void kernel_launch(void* const* d_in, const int* in_sizes, int n_in,
                              void* d_out, int out_size, void* d_ws, size_t ws_size,
                              hipStream_t stream) {
  const float* x   = (const float*)d_in[0];
  const int*   src = (const int*)d_in[1];
  const int*   dst = (const int*)d_in[2];
  const int*   neg = (const int*)d_in[3];
  const float* W0  = (const float*)d_in[4];
  const float* al0 = (const float*)d_in[5];
  const float* ar0 = (const float*)d_in[6];
  const float* b0  = (const float*)d_in[7];
  const float* W1  = (const float*)d_in[8];
  const float* al1 = (const float*)d_in[9];
  const float* ar1 = (const float*)d_in[10];
  const float* b1  = (const float*)d_in[11];
  const float* P1  = (const float*)d_in[12];
  const float* pb1 = (const float*)d_in[13];
  const float* P2  = (const float*)d_in[14];
  const float* pb2 = (const float*)d_in[15];
  const float* P3  = (const float*)d_in[16];
  const float* pb3 = (const float*)d_in[17];
  float* out = (float*)d_out;

  const int DIN = 128, C0 = 256, H0 = 8, H1 = 1, C1 = 32;
  const size_t N = (size_t)in_sizes[0] / DIN;
  const size_t E = (size_t)in_sizes[1];

  // ---- workspace arena (with reuse) ----
  char* w = (char*)d_ws;
  size_t off = 0;
  _Float16* xh   = (_Float16*)(w + off); off = aln(off + N * DIN * 2);
  _Float16* w0t  = (_Float16*)(w + off); off = aln(off + (size_t)C0 * DIN * 2);
  _Float16* w1t  = (_Float16*)(w + off); off = aln(off + (size_t)C1 * C0 * 2);
  size_t f0_off  = off;                  // layer-0 f0 arena; recycled for all of layer 1
  float* f0      = (float*)(w + off);    off = aln(off + N * C0 * 4);
  float* el0     = (float*)(w + off);    off = aln(off + N * H0 * 4);
  float* er0     = (float*)(w + off);    off = aln(off + N * H0 * 4);
  unsigned* m0   = (unsigned*)(w + off); off = aln(off + N * H0 * 4);
  float* den0    = (float*)(w + off);    off = aln(off + N * H0 * 4);
  size_t e0_bytes = E * H0 * 4;
  size_t hb_bytes = N * C0 * 2;
  float* e0      = (float*)(w + off);
  _Float16* hb   = (_Float16*)(w + off); off = aln(off + (e0_bytes > hb_bytes ? e0_bytes : hb_bytes));
  float* out0    = (float*)(w + off);    off = aln(off + N * C0 * 4);

  // layer-1 working set lives inside the (now dead) f0 region
  size_t s = f0_off;
  float* f1    = (float*)(w + s); s = aln(s + N * C1 * 4);
  float* out1  = (float*)(w + s); s = aln(s + N * C1 * 4);
  float* el1   = (float*)(w + s); s = aln(s + N * 4);
  float* er1   = (float*)(w + s); s = aln(s + N * 4);
  unsigned* m1 = (unsigned*)(w + s); s = aln(s + N * 4);
  float* den1  = (float*)(w + s); s = aln(s + N * 4);
  float* e1    = (float*)(w + s); s = aln(s + E * 4);
  (void)ws_size; (void)n_in;

  const int B = 256;

  // ---- layer 0 ----
  hipMemsetAsync(m0,   0, N * H0 * 4, stream);
  hipMemsetAsync(den0, 0, N * H0 * 4, stream);
  hipMemsetAsync(out0, 0, N * C0 * 4, stream);

  k_cvt_x<<<cdiv(N * DIN, B), B, 0, stream>>>(x, xh, (int)(N * DIN));
  k_cvt_w<<<cdiv(C0 * DIN + C1 * C0, B), B, 0, stream>>>(W0, W1, w0t, w1t);

  {
    long long tiles = (long long)cdiv(N, 16) * (C0 / 32);   // 16x32 strips
    k_gemm_wmma<<<cdiv(tiles, 8), B, 0, stream>>>(xh, w0t, f0, (int)N, DIN, C0);
  }
  k_elsum<<<cdiv(N * H0, B), B, 0, stream>>>(f0, al0, ar0, el0, er0, (int)N, H0);
  k_edge_a<<<cdiv(E * H0, B), B, 0, stream>>>(el0, er0, src, dst, e0, m0, (int)E, H0);
  k_edge_b<<<cdiv(E * H0, B), B, 0, stream>>>(e0, m0, den0, dst, (int)E, H0);
  k_edge_c<<<cdiv(E * 32, B), B, 0, stream>>>(e0, den0, f0, src, dst, out0, (int)E, H0);
  k_elu<<<cdiv(N * C0, B), B, 0, stream>>>(out0, b0, hb, (int)(N * C0));

  // ---- layer 1 ----
  hipMemsetAsync(m1,   0, N * 4, stream);
  hipMemsetAsync(den1, 0, N * 4, stream);
  hipMemsetAsync(out1, 0, N * C1 * 4, stream);

  {
    long long tiles = (long long)cdiv(N, 16) * (C1 / 32);   // 16x32 strips
    k_gemm_wmma<<<cdiv(tiles, 8), B, 0, stream>>>(hb, w1t, f1, (int)N, C0, C1);
  }
  k_elsum<<<cdiv(N * H1, B), B, 0, stream>>>(f1, al1, ar1, el1, er1, (int)N, H1);
  k_edge_a<<<cdiv(E * H1, B), B, 0, stream>>>(el1, er1, src, dst, e1, m1, (int)E, H1);
  k_edge_b<<<cdiv(E * H1, B), B, 0, stream>>>(e1, m1, den1, dst, (int)E, H1);
  k_edge_c<<<cdiv(E * 32, B), B, 0, stream>>>(e1, den1, f1, src, dst, out1, (int)E, H1);

  // ---- predictor (b1 folded in) ----
  k_pred<<<cdiv(out_size, 128), 128, 0, stream>>>(out1, b1, P1, pb1, P2, pb2, P3, pb3,
                                                  neg, (int)N, out, out_size);
}
